// GATNet_24498493456720
// MI455X (gfx1250) — matrix operand order repaired and checked by reading, not compiled, not run on hardware
//
#include <hip/hip_runtime.h>
#include <hip/hip_bf16.h>
#include <math.h>

// ---------------------------------------------------------------------------
// 2-layer GAT (GATConv H=8,C=8 -> ELU -> GATConv H=1,C=16 -> log_softmax)
// for MI455X / gfx1250 (wave32, WMMA, async global->LDS).
//
//   * GEMMs: v_wmma_f32_16x16x32_bf16. W1 pre-repacked to exact B-fragment
//     lane layout, staged per-block into LDS via global_load_async_to_lds_b128
//     (ASYNCcnt) and read back with ds_load_b128 -- shared by all 8 waves.
//     A tiles (x rows, f32) are loaded one iteration ahead of the WMMA that
//     consumes them (software pipeline), converted f32->bf16 RNE in registers.
//   * Edge softmax/aggregation via atomics: uint-punned monotone atomicMax
//     for segment max, atomicAdd f32 for denom + message aggregation.
//   * Self-loops folded into node-level kernels.
// ---------------------------------------------------------------------------

#define HEADS 8
#define C1    8
#define C2    16
#define FIN   512
#define HC1   64   // HEADS*C1
#define NEG   0.2f

typedef __attribute__((ext_vector_type(16))) __bf16 v16bf;
typedef __attribute__((ext_vector_type(8)))  float  v8f;

union Frag16 {
    v16bf v;
    unsigned short u[16];
    uint4 q[2];
};

__device__ __forceinline__ unsigned short f2bf(float f) {
    // round-to-nearest-even f32 -> bf16
    unsigned int u = __float_as_uint(f);
    unsigned int r = u + 0x7fffu + ((u >> 16) & 1u);
    return (unsigned short)(r >> 16);
}

// monotone order-preserving float <-> uint for atomicMax-based segment max
__device__ __forceinline__ unsigned int fkey(float f) {
    unsigned int u = __float_as_uint(f);
    return (u & 0x80000000u) ? ~u : (u | 0x80000000u);
}
__device__ __forceinline__ float funkey(unsigned int u) {
    return (u & 0x80000000u) ? __uint_as_float(u & 0x7fffffffu)
                             : __uint_as_float(~u);
}

__device__ __forceinline__ float leaky(float x) { return x > 0.f ? x : NEG * x; }

__device__ __forceinline__ void cvt_a_frag(Frag16& a, float4 f0, float4 f1,
                                           float4 f2, float4 f3) {
    a.u[0]  = f2bf(f0.x); a.u[1]  = f2bf(f0.y); a.u[2]  = f2bf(f0.z); a.u[3]  = f2bf(f0.w);
    a.u[4]  = f2bf(f1.x); a.u[5]  = f2bf(f1.y); a.u[6]  = f2bf(f1.z); a.u[7]  = f2bf(f1.w);
    a.u[8]  = f2bf(f2.x); a.u[9]  = f2bf(f2.y); a.u[10] = f2bf(f2.z); a.u[11] = f2bf(f2.w);
    a.u[12] = f2bf(f3.x); a.u[13] = f2bf(f3.y); a.u[14] = f2bf(f3.z); a.u[15] = f2bf(f3.w);
}

// ---------------------------------------------------------------------------
// Repack W1 [512,64] (row-major, K-major) into bf16 B-fragments.
// Fragment entry ((kc*4 + t)*32 + lane)*16 + j  holds W1[kc*32 + (j<8 ? hi*8 + j : 16 + hi*8 + j-8),
// t*16 + (lane&15)]   (B 32x16 bf16 layout: lanes 0-15 K=0..7,16..23; 16-31 K=8..15,24..31)
// ---------------------------------------------------------------------------
__global__ void repack_w1(const float* __restrict__ W1,
                          unsigned short* __restrict__ bfrag) {
    int idx = blockIdx.x * blockDim.x + threadIdx.x;
    if (idx >= 16 * 4 * 32 * 16) return;
    int j    = idx & 15;
    int lane = (idx >> 4) & 31;
    int t    = (idx >> 9) & 3;
    int kc   = idx >> 11;
    int hi   = lane >> 4, nn = lane & 15;
    int K    = kc * 32 + (j < 8 ? hi * 8 + j : 16 + hi * 8 + (j - 8));
    int col  = t * 16 + nn;
    bfrag[idx] = f2bf(W1[K * HC1 + col]);
}

// Repack W2 [64,16]: 2 K-chunks, 1 column tile.
__global__ void repack_w2(const float* __restrict__ W2,
                          unsigned short* __restrict__ bfrag) {
    int idx = blockIdx.x * blockDim.x + threadIdx.x;
    if (idx >= 2 * 32 * 16) return;
    int j    = idx & 15;
    int lane = (idx >> 4) & 31;
    int kc   = idx >> 9;
    int hi   = lane >> 4, nn = lane & 15;
    int K    = kc * 32 + (j < 8 ? hi * 8 + j : 16 + hi * 8 + (j - 8));
    bfrag[idx] = f2bf(W2[K * C2 + nn]);
}

// ---------------------------------------------------------------------------
// GEMM1: h1[N,64] = x[N,512] @ W1.  One wave = 16-row strip x 64 cols.
// 256 threads = 8 waves = 128 rows per block.  K stepped by 32 (bf16 WMMA).
// Repacked W1 (64KB) staged to LDS with async loads; A pipelined 1 ahead.
// ---------------------------------------------------------------------------
__global__ void __launch_bounds__(256)
gemm1_kernel(const float* __restrict__ x,
             const unsigned short* __restrict__ bfrag,
             float* __restrict__ h1, int n) {
    __shared__ uint4 ldsb[4096];            // 64KB: all B fragments (16 kc x 4 t)
    int tid = threadIdx.x;

    // ---- async stage: global bfrag -> LDS (16 x b128 per thread) ----
    {
        unsigned long long gbase = (unsigned long long)(size_t)bfrag;
        unsigned lds0 = (unsigned)(size_t)(&ldsb[0]);
#pragma unroll
        for (int i = 0; i < 16; ++i) {
            unsigned idx    = (unsigned)(i * 256 + tid);
            unsigned ldsoff = lds0 + idx * 16u;
            unsigned voff   = idx * 16u;
            asm volatile("global_load_async_to_lds_b128 %0, %1, %2"
                         :: "v"(ldsoff), "v"(voff), "s"(gbase) : "memory");
        }
        asm volatile("s_wait_asynccnt 0x0" ::: "memory");
    }
    __syncthreads();

    int wave = tid >> 5;
    int lane = tid & 31;
    int hi   = lane >> 4;
    int lm   = lane & 15;
    int row0 = blockIdx.x * 128 + wave * 16;

    int arow = row0 + lm;
    if (arow >= n) arow = n - 1;            // clamp loads; stores predicated
    const float* xr = x + (size_t)arow * FIN + hi * 8;

    v8f acc[4];
    acc[0] = v8f{}; acc[1] = v8f{}; acc[2] = v8f{}; acc[3] = v8f{};

    // preload A tile for kc = 0
    float4 f0 = reinterpret_cast<const float4*>(xr)[0];
    float4 f1 = reinterpret_cast<const float4*>(xr)[1];
    float4 f2 = reinterpret_cast<const float4*>(xr + 16)[0];
    float4 f3 = reinterpret_cast<const float4*>(xr + 16)[1];

    for (int kc = 0; kc < FIN / 32; ++kc) {
        Frag16 a;
        cvt_a_frag(a, f0, f1, f2, f3);
        if (kc < FIN / 32 - 1) {            // issue next A loads before WMMAs
            const float* xn = xr + (kc + 1) * 32;
            f0 = reinterpret_cast<const float4*>(xn)[0];
            f1 = reinterpret_cast<const float4*>(xn)[1];
            f2 = reinterpret_cast<const float4*>(xn + 16)[0];
            f3 = reinterpret_cast<const float4*>(xn + 16)[1];
        }
#pragma unroll
        for (int t = 0; t < 4; ++t) {
            const uint4* p = &ldsb[(size_t)((kc * 4 + t) * 32 + lane) * 2];
            Frag16 b;
            b.q[0] = p[0];
            b.q[1] = p[1];
            acc[t] = __builtin_amdgcn_wmma_f32_16x16x32_bf16(
                false, a.v, false, b.v, (short)0, acc[t], false, false);
        }
    }

    // D layout: lanes0-15 M=f, lanes16-31 M=f+8, N = lane&15
    float* outp = h1 + (size_t)(row0 + hi * 8) * HC1 + lm;
    if (row0 + 15 < n) {                    // full strip: unpredicated stores
#pragma unroll
        for (int t = 0; t < 4; ++t)
#pragma unroll
            for (int f = 0; f < 8; ++f)
                outp[(size_t)f * HC1 + t * 16] = acc[t][f];
    } else {
#pragma unroll
        for (int t = 0; t < 4; ++t)
#pragma unroll
            for (int f = 0; f < 8; ++f)
                if (row0 + hi * 8 + f < n)
                    outp[(size_t)f * HC1 + t * 16] = acc[t][f];
    }
}

// ---------------------------------------------------------------------------
// GEMM2: h2[N,16] = x2[N,64] @ W2.  2 K-chunks, 1 col tile (B stays in L2).
// ---------------------------------------------------------------------------
__global__ void __launch_bounds__(256)
gemm2_kernel(const float* __restrict__ x2,
             const unsigned short* __restrict__ bfrag,
             float* __restrict__ h2, int n) {
    int wave = threadIdx.x >> 5;
    int lane = threadIdx.x & 31;
    int hi   = lane >> 4;
    int lm   = lane & 15;
    int row0 = blockIdx.x * 128 + wave * 16;

    int arow = row0 + lm;
    if (arow >= n) arow = n - 1;
    const float* xr = x2 + (size_t)arow * HC1 + hi * 8;

    v8f acc = v8f{};
#pragma unroll
    for (int kc = 0; kc < 2; ++kc) {
        const float* xn = xr + kc * 32;
        float4 f0 = reinterpret_cast<const float4*>(xn)[0];
        float4 f1 = reinterpret_cast<const float4*>(xn)[1];
        float4 f2 = reinterpret_cast<const float4*>(xn + 16)[0];
        float4 f3 = reinterpret_cast<const float4*>(xn + 16)[1];
        Frag16 a;
        cvt_a_frag(a, f0, f1, f2, f3);

        const uint4* bq = reinterpret_cast<const uint4*>(bfrag) +
                          (size_t)(kc * 32 + lane) * 2;
        Frag16 b;
        b.q[0] = bq[0];
        b.q[1] = bq[1];
        acc = __builtin_amdgcn_wmma_f32_16x16x32_bf16(
            false, a.v, false, b.v, (short)0, acc, false, false);
    }

    float* outp = h2 + (size_t)(row0 + hi * 8) * C2 + lm;
    if (row0 + 15 < n) {
#pragma unroll
        for (int f = 0; f < 8; ++f) outp[(size_t)f * C2] = acc[f];
    } else {
#pragma unroll
        for (int f = 0; f < 8; ++f)
            if (row0 + hi * 8 + f < n) outp[(size_t)f * C2] = acc[f];
    }
}

// ---------------------------------------------------------------------------
// Layer-1 attention coefficients + state init (self-loop seeds segment max).
// ---------------------------------------------------------------------------
__global__ void attcoef1(const float* __restrict__ h1,
                         const float* __restrict__ att_src,
                         const float* __restrict__ att_dst,
                         float* __restrict__ a_src, float* __restrict__ a_dst,
                         unsigned int* __restrict__ m1u,
                         float* __restrict__ denom1,
                         float* __restrict__ agg1, int n) {
    int nd = blockIdx.x * blockDim.x + threadIdx.x;
    if (nd >= n) return;
    const float* hr = h1 + (size_t)nd * HC1;
#pragma unroll
    for (int h = 0; h < HEADS; ++h) {
        float as = 0.f, ad = 0.f;
#pragma unroll
        for (int c = 0; c < C1; ++c) {
            float v = hr[h * C1 + c];
            as += v * att_src[h * C1 + c];
            ad += v * att_dst[h * C1 + c];
        }
        a_src[(size_t)nd * HEADS + h] = as;
        a_dst[(size_t)nd * HEADS + h] = ad;
        m1u[(size_t)nd * HEADS + h]   = fkey(leaky(as + ad));  // self-loop logit
        denom1[(size_t)nd * HEADS + h] = 0.f;
    }
#pragma unroll
    for (int i = 0; i < HC1; ++i) agg1[(size_t)nd * HC1 + i] = 0.f;
}

__global__ void e1max(const int* __restrict__ ei,
                      const float* __restrict__ a_src,
                      const float* __restrict__ a_dst,
                      unsigned int* __restrict__ m1u, int ne) {
    int e = blockIdx.x * blockDim.x + threadIdx.x;
    if (e >= ne) return;
    int s = ei[e], d = ei[ne + e];
    const float* ps = a_src + (size_t)s * HEADS;
    const float* pd = a_dst + (size_t)d * HEADS;
#pragma unroll
    for (int h = 0; h < HEADS; ++h)
        atomicMax(&m1u[(size_t)d * HEADS + h], fkey(leaky(ps[h] + pd[h])));
}

__global__ void e1sum(const int* __restrict__ ei,
                      const float* __restrict__ a_src,
                      const float* __restrict__ a_dst,
                      const unsigned int* __restrict__ m1u,
                      float* __restrict__ denom1, int ne) {
    int e = blockIdx.x * blockDim.x + threadIdx.x;
    if (e >= ne) return;
    int s = ei[e], d = ei[ne + e];
    const float* ps = a_src + (size_t)s * HEADS;
    const float* pd = a_dst + (size_t)d * HEADS;
#pragma unroll
    for (int h = 0; h < HEADS; ++h) {
        float v = leaky(ps[h] + pd[h]);
        float m = funkey(m1u[(size_t)d * HEADS + h]);
        atomicAdd(&denom1[(size_t)d * HEADS + h], __expf(v - m));
    }
}

__global__ void selfsum1(const float* __restrict__ a_src,
                         const float* __restrict__ a_dst,
                         const unsigned int* __restrict__ m1u,
                         float* __restrict__ denom1, int n) {
    int nd = blockIdx.x * blockDim.x + threadIdx.x;
    if (nd >= n) return;
#pragma unroll
    for (int h = 0; h < HEADS; ++h) {
        size_t i = (size_t)nd * HEADS + h;
        float v = leaky(a_src[i] + a_dst[i]);
        denom1[i] += __expf(v - funkey(m1u[i]));   // kernel-ordered: non-atomic ok
    }
}

__global__ void e1msg(const int* __restrict__ ei,
                      const float* __restrict__ a_src,
                      const float* __restrict__ a_dst,
                      const unsigned int* __restrict__ m1u,
                      const float* __restrict__ denom1,
                      const float* __restrict__ h1,
                      float* __restrict__ agg1, int ne) {
    int e = blockIdx.x * blockDim.x + threadIdx.x;
    if (e >= ne) return;
    int s = ei[e], d = ei[ne + e];
    const float* ps = a_src + (size_t)s * HEADS;
    const float* pd = a_dst + (size_t)d * HEADS;
    const float* hs = h1 + (size_t)s * HC1;
    float* ag = agg1 + (size_t)d * HC1;
#pragma unroll
    for (int h = 0; h < HEADS; ++h) {
        float v = leaky(ps[h] + pd[h]);
        size_t i = (size_t)d * HEADS + h;
        float alpha = __expf(v - funkey(m1u[i])) / (denom1[i] + 1e-16f);
        const float4* hv = reinterpret_cast<const float4*>(hs + h * C1);
        float4 h0 = hv[0], h1v = hv[1];
        atomicAdd(&ag[h * C1 + 0], h0.x * alpha);
        atomicAdd(&ag[h * C1 + 1], h0.y * alpha);
        atomicAdd(&ag[h * C1 + 2], h0.z * alpha);
        atomicAdd(&ag[h * C1 + 3], h0.w * alpha);
        atomicAdd(&ag[h * C1 + 4], h1v.x * alpha);
        atomicAdd(&ag[h * C1 + 5], h1v.y * alpha);
        atomicAdd(&ag[h * C1 + 6], h1v.z * alpha);
        atomicAdd(&ag[h * C1 + 7], h1v.w * alpha);
    }
}

// add self-loop message, bias, ELU -> x2 (written in-place into agg1)
__global__ void final1(const float* __restrict__ h1,
                       const float* __restrict__ a_src,
                       const float* __restrict__ a_dst,
                       const unsigned int* __restrict__ m1u,
                       const float* __restrict__ denom1,
                       const float* __restrict__ b1,
                       float* __restrict__ agg1, int n) {
    int nd = blockIdx.x * blockDim.x + threadIdx.x;
    if (nd >= n) return;
#pragma unroll
    for (int h = 0; h < HEADS; ++h) {
        size_t i = (size_t)nd * HEADS + h;
        float v = leaky(a_src[i] + a_dst[i]);
        float alpha = __expf(v - funkey(m1u[i])) / (denom1[i] + 1e-16f);
#pragma unroll
        for (int c = 0; c < C1; ++c) {
            size_t j = (size_t)nd * HC1 + h * C1 + c;
            float o = agg1[j] + h1[j] * alpha + b1[h * C1 + c];
            agg1[j] = o > 0.f ? o : __expf(o) - 1.f;   // ELU
        }
    }
}

// ---------------------------------------------------------------------------
// Layer 2 (H=1, C=16)
// ---------------------------------------------------------------------------
__global__ void attcoef2(const float* __restrict__ h2,
                         const float* __restrict__ att_src,
                         const float* __restrict__ att_dst,
                         float* __restrict__ a_src, float* __restrict__ a_dst,
                         unsigned int* __restrict__ m2u,
                         float* __restrict__ denom2,
                         float* __restrict__ agg2, int n) {
    int nd = blockIdx.x * blockDim.x + threadIdx.x;
    if (nd >= n) return;
    const float* hr = h2 + (size_t)nd * C2;
    float as = 0.f, ad = 0.f;
#pragma unroll
    for (int c = 0; c < C2; ++c) {
        as += hr[c] * att_src[c];
        ad += hr[c] * att_dst[c];
    }
    a_src[nd] = as;
    a_dst[nd] = ad;
    m2u[nd]   = fkey(leaky(as + ad));
    denom2[nd] = 0.f;
#pragma unroll
    for (int c = 0; c < C2; ++c) agg2[(size_t)nd * C2 + c] = 0.f;
}

__global__ void e2max(const int* __restrict__ ei,
                      const float* __restrict__ a_src,
                      const float* __restrict__ a_dst,
                      unsigned int* __restrict__ m2u, int ne) {
    int e = blockIdx.x * blockDim.x + threadIdx.x;
    if (e >= ne) return;
    int s = ei[e], d = ei[ne + e];
    atomicMax(&m2u[d], fkey(leaky(a_src[s] + a_dst[d])));
}

__global__ void e2sum(const int* __restrict__ ei,
                      const float* __restrict__ a_src,
                      const float* __restrict__ a_dst,
                      const unsigned int* __restrict__ m2u,
                      float* __restrict__ denom2, int ne) {
    int e = blockIdx.x * blockDim.x + threadIdx.x;
    if (e >= ne) return;
    int s = ei[e], d = ei[ne + e];
    float v = leaky(a_src[s] + a_dst[d]);
    atomicAdd(&denom2[d], __expf(v - funkey(m2u[d])));
}

__global__ void selfsum2(const float* __restrict__ a_src,
                         const float* __restrict__ a_dst,
                         const unsigned int* __restrict__ m2u,
                         float* __restrict__ denom2, int n) {
    int nd = blockIdx.x * blockDim.x + threadIdx.x;
    if (nd >= n) return;
    float v = leaky(a_src[nd] + a_dst[nd]);
    denom2[nd] += __expf(v - funkey(m2u[nd]));
}

__global__ void e2msg(const int* __restrict__ ei,
                      const float* __restrict__ a_src,
                      const float* __restrict__ a_dst,
                      const unsigned int* __restrict__ m2u,
                      const float* __restrict__ denom2,
                      const float* __restrict__ h2,
                      float* __restrict__ agg2, int ne) {
    int e = blockIdx.x * blockDim.x + threadIdx.x;
    if (e >= ne) return;
    int s = ei[e], d = ei[ne + e];
    float v = leaky(a_src[s] + a_dst[d]);
    float alpha = __expf(v - funkey(m2u[d])) / (denom2[d] + 1e-16f);
    const float* hs = h2 + (size_t)s * C2;
    float* ag = agg2 + (size_t)d * C2;
#pragma unroll
    for (int c = 0; c < C2; ++c) atomicAdd(&ag[c], hs[c] * alpha);
}

__global__ void final2(const float* __restrict__ h2,
                       const float* __restrict__ a_src,
                       const float* __restrict__ a_dst,
                       const unsigned int* __restrict__ m2u,
                       const float* __restrict__ denom2,
                       const float* __restrict__ b2,
                       const float* __restrict__ agg2,
                       float* __restrict__ out, int n) {
    int nd = blockIdx.x * blockDim.x + threadIdx.x;
    if (nd >= n) return;
    float e = leaky(a_src[nd] + a_dst[nd]);
    float alpha = __expf(e - funkey(m2u[nd])) / (denom2[nd] + 1e-16f);
    float v[C2];
    float mx = -1e30f;
#pragma unroll
    for (int c = 0; c < C2; ++c) {
        v[c] = agg2[(size_t)nd * C2 + c] + h2[(size_t)nd * C2 + c] * alpha + b2[c];
        mx = fmaxf(mx, v[c]);
    }
    float sum = 0.f;
#pragma unroll
    for (int c = 0; c < C2; ++c) sum += __expf(v[c] - mx);
    float lse = __logf(sum);
#pragma unroll
    for (int c = 0; c < C2; ++c) out[(size_t)nd * C2 + c] = v[c] - mx - lse;
}

// ---------------------------------------------------------------------------
extern "C" void kernel_launch(void* const* d_in, const int* in_sizes, int n_in,
                              void* d_out, int out_size, void* d_ws, size_t ws_size,
                              hipStream_t stream) {
    const float* x     = (const float*)d_in[0];
    const int*   ei    = (const int*)d_in[1];
    const float* W1    = (const float*)d_in[2];
    const float* as1   = (const float*)d_in[3];
    const float* ad1   = (const float*)d_in[4];
    const float* b1    = (const float*)d_in[5];
    const float* W2    = (const float*)d_in[6];
    const float* as2   = (const float*)d_in[7];
    const float* ad2   = (const float*)d_in[8];
    const float* b2    = (const float*)d_in[9];
    float* out = (float*)d_out;

    const int n  = in_sizes[0] / FIN;   // 100000
    const int ne = in_sizes[1] / 2;     // 1600000

    // ---- workspace layout (256B aligned); layer-2 buffers overlay dead h1 ----
    char* ws = (char*)d_ws;
    size_t off = 0;
    auto alloc = [&](size_t bytes) {
        size_t o = off;
        off = (off + bytes + 255) & ~(size_t)255;
        return o;
    };
    size_t o_bf1   = alloc(16 * 4 * 32 * 16 * sizeof(unsigned short)); // 64KB
    size_t o_bf2   = alloc(2 * 32 * 16 * sizeof(unsigned short));      // 2KB
    size_t o_h1    = alloc((size_t)n * HC1 * 4);
    size_t o_asrc1 = alloc((size_t)n * HEADS * 4);
    size_t o_adst1 = alloc((size_t)n * HEADS * 4);
    size_t o_m1    = alloc((size_t)n * HEADS * 4);
    size_t o_den1  = alloc((size_t)n * HEADS * 4);
    size_t o_agg1  = alloc((size_t)n * HC1 * 4);
    // layer-2 region reuses h1 span (h1 dead after final1; gemm2 reads agg1)
    size_t l2 = o_h1;
    size_t o_h2    = l2;               l2 += ((size_t)n * C2 * 4 + 255) & ~(size_t)255;
    size_t o_asrc2 = l2;               l2 += ((size_t)n * 4 + 255) & ~(size_t)255;
    size_t o_adst2 = l2;               l2 += ((size_t)n * 4 + 255) & ~(size_t)255;
    size_t o_m2    = l2;               l2 += ((size_t)n * 4 + 255) & ~(size_t)255;
    size_t o_den2  = l2;               l2 += ((size_t)n * 4 + 255) & ~(size_t)255;
    size_t o_agg2  = l2;

    unsigned short* bf1 = (unsigned short*)(ws + o_bf1);
    unsigned short* bf2 = (unsigned short*)(ws + o_bf2);
    float* h1    = (float*)(ws + o_h1);
    float* asrc1 = (float*)(ws + o_asrc1);
    float* adst1 = (float*)(ws + o_adst1);
    unsigned int* m1u = (unsigned int*)(ws + o_m1);
    float* den1  = (float*)(ws + o_den1);
    float* agg1  = (float*)(ws + o_agg1);   // becomes x2 after final1
    float* h2    = (float*)(ws + o_h2);
    float* asrc2 = (float*)(ws + o_asrc2);
    float* adst2 = (float*)(ws + o_adst2);
    unsigned int* m2u = (unsigned int*)(ws + o_m2);
    float* den2  = (float*)(ws + o_den2);
    float* agg2  = (float*)(ws + o_agg2);

    const int TB = 256;
    int gn = (n + TB - 1) / TB;
    int ge = (ne + TB - 1) / TB;
    int gg = (n + 127) / 128;

    // ---- weights -> bf16 B-fragments (lives in L2 for the whole grid) ----
    repack_w1<<<(32768 + TB - 1) / TB, TB, 0, stream>>>(W1, bf1);
    repack_w2<<<(1024 + TB - 1) / TB, TB, 0, stream>>>(W2, bf2);

    // ---- layer 1 ----
    gemm1_kernel<<<gg, TB, 0, stream>>>(x, bf1, h1, n);
    attcoef1<<<gn, TB, 0, stream>>>(h1, as1, ad1, asrc1, adst1, m1u, den1, agg1, n);
    e1max<<<ge, TB, 0, stream>>>(ei, asrc1, adst1, m1u, ne);
    e1sum<<<ge, TB, 0, stream>>>(ei, asrc1, adst1, m1u, den1, ne);
    selfsum1<<<gn, TB, 0, stream>>>(asrc1, adst1, m1u, den1, n);
    e1msg<<<ge, TB, 0, stream>>>(ei, asrc1, adst1, m1u, den1, h1, agg1, ne);
    final1<<<gn, TB, 0, stream>>>(h1, asrc1, adst1, m1u, den1, b1, agg1, n);

    // ---- layer 2 ----
    gemm2_kernel<<<gg, TB, 0, stream>>>(agg1, bf2, h2, n);
    attcoef2<<<gn, TB, 0, stream>>>(h2, as2, ad2, asrc2, adst2, m2u, den2, agg2, n);
    e2max<<<ge, TB, 0, stream>>>(ei, asrc2, adst2, m2u, ne);
    e2sum<<<ge, TB, 0, stream>>>(ei, asrc2, adst2, m2u, den2, ne);
    selfsum2<<<gn, TB, 0, stream>>>(asrc2, adst2, m2u, den2, n);
    e2msg<<<ge, TB, 0, stream>>>(ei, asrc2, adst2, m2u, den2, h2, agg2, ne);
    final2<<<gn, TB, 0, stream>>>(h2, asrc2, adst2, m2u, den2, b2, agg2, out, n);

    (void)n_in; (void)out_size; (void)ws_size;
}
